// TransformerEncoderLayer_58548994179296
// MI455X (gfx1250) — matrix operand order repaired and checked by reading, not compiled, not run on hardware
//
#include <hip/hip_runtime.h>

// ---------------------------------------------------------------------------
// Transformer encoder layer for MI455X (gfx1250), f16 WMMA + f32 accumulate.
// B=4 S=2048 D=1024 H=16 DH=64 F=4096
// - GEMM tiles staged with GLOBAL_LOAD_ASYNC_TO_LDS_B128 (ASYNCcnt),
//   double-buffered so the async DMA overlaps the WMMA pipeline.
// - Flash-attention softmax reductions via v_permlane16_b32 (pure VALU,
//   no LDS round-trips).
// ---------------------------------------------------------------------------

typedef _Float16 f16;
typedef __attribute__((ext_vector_type(16))) _Float16 v16h;
typedef __attribute__((ext_vector_type(8)))  _Float16 v8h;
typedef __attribute__((ext_vector_type(8)))  float    v8f;

union F16Frag { v16h v; v8h h[2]; };

__device__ __forceinline__ v8f wmma16x16x32(v16h a, v16h b, v8f c) {
  return __builtin_amdgcn_wmma_f32_16x16x32_f16(false, a, false, b, (short)0, c,
                                                false, false);
}

// Async 16B copy global -> LDS. Generic pointers to __shared__ carry the LDS
// byte offset in their low 32 bits (flat aperture encoding, ISA 10.2), which
// is what the VDST operand of the async-to-LDS ops expects.
__device__ __forceinline__ void async_ld_b128(void* lds_dst, const void* g) {
  uint32_t off = (uint32_t)(uintptr_t)lds_dst;
  asm volatile("global_load_async_to_lds_b128 %0, %1, off"
               :: "v"(off), "v"(g) : "memory");
}
__device__ __forceinline__ void wait_async0() {
  asm volatile("s_wait_asynccnt 0x0" ::: "memory");
}

// Cross-lane permute within each 16-lane row (VALU, no LDS).
__device__ __forceinline__ float permrow16(float v, int sel0, int sel1) {
  int i = __builtin_bit_cast(int, v);
  int r = __builtin_amdgcn_permlane16(i, i, sel0, sel1, false, false);
  return __builtin_bit_cast(float, r);
}
// max / sum over the 16 lanes of each half-wave row.
__device__ __forceinline__ float rowmax16(float v) {
  v = fmaxf(v, permrow16(v, 0x67452301, 0xEFCDAB89));  // xor 1
  v = fmaxf(v, permrow16(v, 0x54761032, 0xDCFE98BA));  // xor 2
  v = fmaxf(v, permrow16(v, 0x32107654, 0xBA98FEDC));  // xor 4
  v = fmaxf(v, permrow16(v, 0xFEDCBA98, 0x76543210));  // xor 8
  return v;
}
__device__ __forceinline__ float rowsum16(float v) {
  v += permrow16(v, 0x67452301, 0xEFCDAB89);
  v += permrow16(v, 0x54761032, 0xDCFE98BA);
  v += permrow16(v, 0x32107654, 0xBA98FEDC);
  v += permrow16(v, 0xFEDCBA98, 0x76543210);
  return v;
}

constexpr int Bdim = 4, SEQ = 2048, DM = 1024, NH = 16, HD = 64, FF = 4096;
constexpr int MT = Bdim * SEQ;  // 8192 rows

// ---------------------------------------------------------------------------
// Prep kernels
// ---------------------------------------------------------------------------
__global__ void cvt_f32_f16_kernel(const float* __restrict__ in,
                                   f16* __restrict__ out, size_t n) {
  size_t i = (size_t)blockIdx.x * blockDim.x + threadIdx.x;
  if (i < n) out[i] = (f16)in[i];
}

// out[n*K + k] = in[k*N + n]  (row-major [K,N] f32 -> BT [N,K] f16)
__global__ void transpose_f32_f16_kernel(const float* __restrict__ in,
                                         f16* __restrict__ out, int K, int N) {
  size_t i = (size_t)blockIdx.x * blockDim.x + threadIdx.x;
  if (i < (size_t)K * N) {
    int n = (int)(i / K);
    int k = (int)(i % K);
    out[i] = (f16)in[(size_t)k * N + n];
  }
}

// Wq is [H, D, DH]; effective B matrix is [K=D][N=H*DH].
__global__ void prep_wqkvT_kernel(const float* __restrict__ W,
                                  f16* __restrict__ wt) {
  size_t i = (size_t)blockIdx.x * blockDim.x + threadIdx.x;  // N*D = 1M
  if (i < (size_t)DM * DM) {
    int n = (int)(i >> 10);
    int k = (int)(i & 1023);
    wt[i] = (f16)W[((size_t)((n >> 6) * DM + k)) * HD + (n & 63)];
  }
}

// ---------------------------------------------------------------------------
// Tiled GEMM:  C[M,N] = A[M,K] (f16 rm) * BT[N,K] (f16) + bias
// MODE 0: f16 row-major out.  MODE 1: f16 transposed out (out[n*ldo+m]).
// MODE 2: f32 out = acc + bias + res[m*ldo+n].
// Double-buffered async-to-LDS staging.
// ---------------------------------------------------------------------------
template <int MODE, bool RELU>
__global__ __launch_bounds__(256) void gemm_f16_kernel(
    const f16* __restrict__ A, int lda, const f16* __restrict__ BT, int ldb,
    void* __restrict__ out, int ldo, const float* __restrict__ bias,
    const float* __restrict__ res, int M, int N, int K) {
  constexpr int BM = 128, BN = 128, BK = 32, LD = BK + 8;  // pad: 80B stride
  __shared__ __align__(16) f16 As[2][BM * LD];
  __shared__ __align__(16) f16 Bs[2][BN * LD];

  const int tid = threadIdx.x;
  const int lane = tid & 31;
  const int wave = tid >> 5;       // 8 waves: 2 (M) x 4 (N)
  const int wm0 = (wave & 1) * 64; // wave tile 64x32
  const int wn0 = (wave >> 1) * 32;
  const int bm0 = blockIdx.x * BM;
  const int bn0 = blockIdx.y * BN;
  const int l15 = lane & 15;
  const int half = lane >> 4;

  v8f acc[4][2] = {};

  const int grow = tid >> 1;        // 0..127
  const int gcol = (tid & 1) * 16;  // 0 or 16
  const f16* ag = A + (size_t)(bm0 + grow) * lda + gcol;
  const f16* bg = BT + (size_t)(bn0 + grow) * ldb + gcol;
  const int ldst = grow * LD + gcol;

  auto issue = [&](int buf, int k0) {
    async_ld_b128(&As[buf][ldst],     ag + k0);
    async_ld_b128(&As[buf][ldst + 8], ag + k0 + 8);
    async_ld_b128(&Bs[buf][ldst],     bg + k0);
    async_ld_b128(&Bs[buf][ldst + 8], bg + k0 + 8);
  };

  issue(0, 0);
  const int nk = K / BK;
  for (int kt = 0; kt < nk; ++kt) {
    wait_async0();
    __syncthreads();
    if (kt + 1 < nk) issue((kt + 1) & 1, (kt + 1) * BK);
    const int cur = kt & 1;

    F16Frag a[4], b[2];
    const int kca = half * 8;   // A layout: {kca..kca+7, kca+16..kca+23}
    const int kcb = half * 8;   // B layout: {kcb..kcb+7, kcb+16..kcb+23}
#pragma unroll
    for (int mt = 0; mt < 4; ++mt) {
      const f16* p = &As[cur][(wm0 + mt * 16 + l15) * LD + kca];
      a[mt].h[0] = *(const v8h*)p;
      a[mt].h[1] = *(const v8h*)(p + 16);
    }
#pragma unroll
    for (int nt = 0; nt < 2; ++nt) {
      const f16* p = &Bs[cur][(wn0 + nt * 16 + l15) * LD + kcb];
      b[nt].h[0] = *(const v8h*)p;
      b[nt].h[1] = *(const v8h*)(p + 16);
    }
#pragma unroll
    for (int mt = 0; mt < 4; ++mt)
#pragma unroll
      for (int nt = 0; nt < 2; ++nt)
        acc[mt][nt] = wmma16x16x32(a[mt].v, b[nt].v, acc[mt][nt]);
  }

  // Epilogue.  C layout: lane holds N = l15, M = r + 8*half (r = 0..7).
#pragma unroll
  for (int mt = 0; mt < 4; ++mt) {
#pragma unroll
    for (int nt = 0; nt < 2; ++nt) {
      const int gn = bn0 + wn0 + nt * 16 + l15;
      const float bv = bias[gn];
      float vals[8];
#pragma unroll
      for (int r = 0; r < 8; ++r) {
        float v = acc[mt][nt][r] + bv;
        if (RELU) v = v > 0.f ? v : 0.f;
        vals[r] = v;
      }
      const int gmb = bm0 + wm0 + mt * 16 + half * 8;
      if (MODE == 0) {
        f16* o = (f16*)out;
#pragma unroll
        for (int r = 0; r < 8; ++r)
          o[(size_t)(gmb + r) * ldo + gn] = (f16)vals[r];
      } else if (MODE == 1) {
        f16* o = (f16*)out;
        v8h hv;
#pragma unroll
        for (int r = 0; r < 8; ++r) hv[r] = (f16)vals[r];
        *(v8h*)&o[(size_t)gn * ldo + gmb] = hv;  // contiguous 16B store
      } else {
        float* o = (float*)out;
#pragma unroll
        for (int r = 0; r < 8; ++r) {
          size_t idx = (size_t)(gmb + r) * ldo + gn;
          o[idx] = vals[r] + res[idx];
        }
      }
    }
  }
}

// ---------------------------------------------------------------------------
// Flash attention. Q,K: [B*S, DM] f16 (head h at col h*64). Vt: [H*HD, MT].
// Grid: (B*H, S/128); 8 waves/block, each owns a 16-query tile, 64-key blocks.
// ---------------------------------------------------------------------------
__global__ __launch_bounds__(256) void attention_kernel(
    const f16* __restrict__ Q, const f16* __restrict__ Kc,
    const f16* __restrict__ Vt, f16* __restrict__ Ctx) {
  constexpr int KB = 64;
  const int b = blockIdx.x / NH, h = blockIdx.x % NH;
  const int wave = threadIdx.x >> 5, lane = threadIdx.x & 31;
  const int l15 = lane & 15, half = lane >> 4;
  const int q0 = blockIdx.y * 128 + wave * 16;

  __shared__ __align__(16) f16 Ks[KB][HD + 8];   // [key][e]
  __shared__ __align__(16) f16 Vs[HD][KB + 8];   // [e][key]
  __shared__ __align__(16) f16 Ps[8][16][KB];    // per-wave P tile

  // Q fragments (A layout), 2 K-steps over e = 0..63, kept in registers.
  F16Frag qf[2];
  const f16* qp = Q + (size_t)(b * SEQ + q0 + l15) * DM + h * HD;
#pragma unroll
  for (int e = 0; e < 2; ++e) {
    const int kc = e * 32 + half * 8;
    qf[e].h[0] = *(const v8h*)(qp + kc);
    qf[e].h[1] = *(const v8h*)(qp + kc + 16);
  }

  float rmax[8], rsum[8];
#pragma unroll
  for (int r = 0; r < 8; ++r) { rmax[r] = -3.0e38f; rsum[r] = 0.f; }
  v8f o[4] = {};

  const int srow = threadIdx.x >> 2;        // 0..63
  const int scol = (threadIdx.x & 3) * 16;  // 0,16,32,48

  for (int kb0 = 0; kb0 < SEQ; kb0 += KB) {
    __syncthreads();  // everyone done reading previous K/V tiles
    {  // async stage of K tile and V^T tile (shared by all 8 waves)
      const f16* kg = Kc + (size_t)(b * SEQ + kb0 + srow) * DM + h * HD + scol;
      async_ld_b128(&Ks[srow][scol],     kg);
      async_ld_b128(&Ks[srow][scol + 8], kg + 8);
      const f16* vg = Vt + (size_t)(h * HD + srow) * MT + b * SEQ + kb0 + scol;
      async_ld_b128(&Vs[srow][scol],     vg);
      async_ld_b128(&Vs[srow][scol + 8], vg + 8);
    }
    wait_async0();
    __syncthreads();

    // scores: 4 key n-tiles x 2 e-steps
    v8f s[4] = {};
#pragma unroll
    for (int nt = 0; nt < 4; ++nt) {
#pragma unroll
      for (int e = 0; e < 2; ++e) {
        F16Frag kf;
        const f16* p = &Ks[nt * 16 + l15][e * 32 + half * 8];
        kf.h[0] = *(const v8h*)p;
        kf.h[1] = *(const v8h*)(p + 16);
        s[nt] = wmma16x16x32(qf[e].v, kf.v, s[nt]);
      }
    }
#pragma unroll
    for (int nt = 0; nt < 4; ++nt)
#pragma unroll
      for (int r = 0; r < 8; ++r) s[nt][r] *= 0.125f;  // 1/sqrt(64)

    // online softmax; row m = r + 8*half; stats uniform across each 16-lane half
    float corr[8];
#pragma unroll
    for (int r = 0; r < 8; ++r) {
      float mx = s[0][r];
#pragma unroll
      for (int nt = 1; nt < 4; ++nt) mx = fmaxf(mx, s[nt][r]);
      mx = rowmax16(mx);
      float nm = fmaxf(rmax[r], mx);
      corr[r] = __expf(rmax[r] - nm);
      rmax[r] = nm;
      float ps = 0.f;
#pragma unroll
      for (int nt = 0; nt < 4; ++nt) {
        float pv = __expf(s[nt][r] - nm);
        s[nt][r] = pv;
        ps += pv;
      }
      ps = rowsum16(ps);
      rsum[r] = rsum[r] * corr[r] + ps;
    }
#pragma unroll
    for (int t = 0; t < 4; ++t)
#pragma unroll
      for (int r = 0; r < 8; ++r) o[t][r] *= corr[r];

    // transpose P through per-wave LDS into A-fragment layout
#pragma unroll
    for (int nt = 0; nt < 4; ++nt)
#pragma unroll
      for (int r = 0; r < 8; ++r)
        Ps[wave][r + 8 * half][nt * 16 + l15] = (f16)s[nt][r];
    asm volatile("s_wait_dscnt 0x0" ::: "memory");

    // ctx += P @ V : 2 key K-steps x 4 e n-tiles
#pragma unroll
    for (int kk = 0; kk < 2; ++kk) {
      F16Frag pf;
      const f16* pp = &Ps[wave][l15][kk * 32 + half * 8];
      pf.h[0] = *(const v8h*)pp;
      pf.h[1] = *(const v8h*)(pp + 16);
#pragma unroll
      for (int t = 0; t < 4; ++t) {
        F16Frag vf;
        const f16* vp = &Vs[t * 16 + l15][kk * 32 + half * 8];
        vf.h[0] = *(const v8h*)vp;
        vf.h[1] = *(const v8h*)(vp + 16);
        o[t] = wmma16x16x32(pf.v, vf.v, o[t]);
      }
    }
  }

  // normalize and store (concat heads): ctx[(b*S+q), h*64+e]
#pragma unroll
  for (int t = 0; t < 4; ++t)
#pragma unroll
    for (int r = 0; r < 8; ++r) {
      float v = o[t][r] / rsum[r];
      size_t idx = (size_t)(b * SEQ + q0 + r + 8 * half) * DM + h * HD +
                   t * 16 + l15;
      Ctx[idx] = (f16)v;
    }
}

// ---------------------------------------------------------------------------
// LayerNorm (one row per block), optional f16 shadow copy for next GEMM.
// ---------------------------------------------------------------------------
__global__ __launch_bounds__(256) void layernorm_kernel(
    const float* __restrict__ y, const float* __restrict__ w,
    const float* __restrict__ bb, float* __restrict__ xo,
    f16* __restrict__ xh, int D) {
  __shared__ float red[256];
  const int tid = threadIdx.x;
  const float* yr = y + (size_t)blockIdx.x * D;
  float s = 0.f;
  for (int i = tid; i < D; i += 256) s += yr[i];
  red[tid] = s;
  __syncthreads();
  for (int st = 128; st > 0; st >>= 1) {
    if (tid < st) red[tid] += red[tid + st];
    __syncthreads();
  }
  const float mean = red[0] / D;
  __syncthreads();
  float vs = 0.f;
  for (int i = tid; i < D; i += 256) {
    float d = yr[i] - mean;
    vs += d * d;
  }
  red[tid] = vs;
  __syncthreads();
  for (int st = 128; st > 0; st >>= 1) {
    if (tid < st) red[tid] += red[tid + st];
    __syncthreads();
  }
  const float rstd = rsqrtf(red[0] / D + 1e-5f);
  for (int i = tid; i < D; i += 256) {
    float v = (yr[i] - mean) * rstd * w[i] + bb[i];
    xo[(size_t)blockIdx.x * D + i] = v;
    if (xh) xh[(size_t)blockIdx.x * D + i] = (f16)v;
  }
}

// ---------------------------------------------------------------------------
extern "C" void kernel_launch(void* const* d_in, const int* in_sizes, int n_in,
                              void* d_out, int out_size, void* d_ws,
                              size_t ws_size, hipStream_t stream) {
  const float* src  = (const float*)d_in[0];
  const float* Wq   = (const float*)d_in[1];
  const float* bq   = (const float*)d_in[2];
  const float* Wk   = (const float*)d_in[3];
  const float* bk   = (const float*)d_in[4];
  const float* Wv   = (const float*)d_in[5];
  const float* bv   = (const float*)d_in[6];
  const float* Wo   = (const float*)d_in[7];
  const float* bo   = (const float*)d_in[8];
  const float* ln1w = (const float*)d_in[9];
  const float* ln1b = (const float*)d_in[10];
  const float* W1   = (const float*)d_in[11];
  const float* b1   = (const float*)d_in[12];
  const float* W2   = (const float*)d_in[13];
  const float* b2   = (const float*)d_in[14];
  const float* ln2w = (const float*)d_in[15];
  const float* ln2b = (const float*)d_in[16];

  char* p = (char*)d_ws;
  auto alloc = [&](size_t bytes) {
    void* r = (void*)p;
    p += (bytes + 255) & ~(size_t)255;
    return r;
  };
  f16*   srcH = (f16*)alloc((size_t)MT * DM * 2);   // also reused as xh
  f16*   qh   = (f16*)alloc((size_t)MT * DM * 2);
  f16*   kh   = (f16*)alloc((size_t)MT * DM * 2);
  f16*   vt   = (f16*)alloc((size_t)MT * DM * 2);   // [H*HD, MT]
  f16*   ctx  = (f16*)alloc((size_t)MT * DM * 2);
  f16*   WqT  = (f16*)alloc((size_t)DM * DM * 2);
  f16*   WkT  = (f16*)alloc((size_t)DM * DM * 2);
  f16*   WvT  = (f16*)alloc((size_t)DM * DM * 2);
  f16*   WoT  = (f16*)alloc((size_t)DM * DM * 2);
  f16*   W1T  = (f16*)alloc((size_t)DM * FF * 2);
  f16*   W2T  = (f16*)alloc((size_t)DM * FF * 2);
  float* y    = (float*)alloc((size_t)MT * DM * 4); // reused for FFN2 out
  float* x    = (float*)alloc((size_t)MT * DM * 4);
  f16*   hb   = (f16*)alloc((size_t)MT * FF * 2);

  const size_t nSrc = (size_t)MT * DM;
  cvt_f32_f16_kernel<<<(nSrc + 255) / 256, 256, 0, stream>>>(src, srcH, nSrc);
  prep_wqkvT_kernel<<<(DM * DM + 255) / 256, 256, 0, stream>>>(Wq, WqT);
  prep_wqkvT_kernel<<<(DM * DM + 255) / 256, 256, 0, stream>>>(Wk, WkT);
  prep_wqkvT_kernel<<<(DM * DM + 255) / 256, 256, 0, stream>>>(Wv, WvT);
  transpose_f32_f16_kernel<<<(DM * DM + 255) / 256, 256, 0, stream>>>(Wo, WoT, DM, DM);
  transpose_f32_f16_kernel<<<(DM * FF + 255) / 256, 256, 0, stream>>>(W1, W1T, DM, FF);
  transpose_f32_f16_kernel<<<(DM * FF + 255) / 256, 256, 0, stream>>>(W2, W2T, FF, DM);

  // Q, K projections (f16 row-major); V projection stored transposed.
  dim3 gQ(MT / 128, DM / 128);
  gemm_f16_kernel<0, false><<<gQ, 256, 0, stream>>>(srcH, DM, WqT, DM, qh, DM,
                                                    bq, nullptr, MT, DM, DM);
  gemm_f16_kernel<0, false><<<gQ, 256, 0, stream>>>(srcH, DM, WkT, DM, kh, DM,
                                                    bk, nullptr, MT, DM, DM);
  gemm_f16_kernel<1, false><<<gQ, 256, 0, stream>>>(srcH, DM, WvT, DM, vt, MT,
                                                    bv, nullptr, MT, DM, DM);

  attention_kernel<<<dim3(Bdim * NH, SEQ / 128), 256, 0, stream>>>(qh, kh, vt, ctx);

  // out-proj + bias + residual(src) -> y (f32)
  gemm_f16_kernel<2, false><<<gQ, 256, 0, stream>>>(ctx, DM, WoT, DM, y, DM,
                                                    bo, src, MT, DM, DM);
  layernorm_kernel<<<MT, 256, 0, stream>>>(y, ln1w, ln1b, x, srcH, DM);

  // FFN1 (relu, f16 out), FFN2 (+ bias + residual(x) -> y), LN2 -> d_out
  gemm_f16_kernel<0, true><<<dim3(MT / 128, FF / 128), 256, 0, stream>>>(
      srcH, DM, W1T, DM, hb, FF, b1, nullptr, MT, FF, DM);
  gemm_f16_kernel<2, false><<<gQ, 256, 0, stream>>>(hb, FF, W2T, FF, y, DM,
                                                    b2, x, MT, DM, FF);
  layernorm_kernel<<<MT, 256, 0, stream>>>(y, ln2w, ln2b, (float*)d_out,
                                           nullptr, DM);
}
